// AdaCoFNet_58703613002438
// MI455X (gfx1250) — compile-verified, matrix operand order checked by reading, not run on hardware
//
#include <hip/hip_runtime.h>
#include <hip/hip_bf16.h>

// ---------------------------------------------------------------------------
// AdaCoF-Net for MI455X (gfx1250): implicit-GEMM conv3x3 via
// v_wmma_f32_16x16x32_bf16, async global->LDS weight staging (ASYNCcnt),
// double-buffered LDS, division-free K loop, VALU tail kernels.
// ---------------------------------------------------------------------------

typedef __attribute__((ext_vector_type(16))) __bf16 v16bf;
typedef __attribute__((ext_vector_type(8)))  float  v8f;

union Frag32 { uint4 u[2]; v16bf v; };

constexpr int Himg = 256, Wimg = 256, BATCHN = 2;
constexpr int HPc = 258, WPc = 258;      // spatially padded activation planes
constexpr int CPc = 288;                 // 266 input channels padded to x32
constexpr int HWi = Himg * Wimg;

__device__ __forceinline__ __bf16 f2bf(float f) {
    union { float f; unsigned u; } cv; cv.f = f;
    unsigned u = cv.u;
    unsigned r = u + 0x7FFFu + ((u >> 16) & 1u);   // round-to-nearest-even
    unsigned short h = (unsigned short)(r >> 16);
    union { unsigned short s; __bf16 b; } cb; cb.s = h;
    return cb.b;
}

// ---------------------------------------------------------------------------
// Weight packing: OIHW f32 -> per-lane WMMA B-fragment bf16 layout.
// dst index t = ((nt*nChunks)+c)*32 + lane ; each t owns 16 bf16 (32B).
// K index = (ky*3+kx)*CP + ci ; chunk c covers K in [c*32, c*32+31].
// ---------------------------------------------------------------------------
__global__ void pack_weights_k(const float* __restrict__ Wsrc, __bf16* __restrict__ dst,
                               int O, int I, int CP, int nChunks, int total) {
    int t = blockIdx.x * blockDim.x + threadIdx.x;
    if (t >= total) return;
    int lane = t & 31;
    int co = ((t >> 5) / nChunks) * 16 + (lane & 15);
    int kb = ((t >> 5) % nChunks) * 32 + (lane >> 4) * 8;
    __bf16* out = dst + (size_t)t * 16;
#pragma unroll
    for (int j = 0; j < 16; ++j) {
        int k = (j < 8) ? (kb + j) : (kb + 16 + (j - 8));
        int pos = k / CP, ci = k % CP;
        float v = 0.f;
        if (co < O && ci < I)
            v = Wsrc[(((size_t)co * I + ci) * 3 + pos / 3) * 3 + (pos % 3)];
        out[j] = f2bf(v);
    }
}

// ---------------------------------------------------------------------------
// Small direct conv for the frame feature extractor (3->16, 16->16) + lrelu.
// ---------------------------------------------------------------------------
template<int CIN>
__global__ void frconv_k(const float* __restrict__ in, const float* __restrict__ wt,
                         const float* __restrict__ bias, float* __restrict__ out) {
    int idx = blockIdx.x * blockDim.x + threadIdx.x;
    if (idx >= BATCHN * HWi) return;
    int b = idx / HWi;
    int rp = idx % HWi;
    int y = rp / Wimg, x = rp % Wimg;
    float acc[16];
#pragma unroll
    for (int co = 0; co < 16; ++co) acc[co] = bias[co];
    for (int ci = 0; ci < CIN; ++ci) {
        const float* ip = in + ((size_t)b * CIN + ci) * HWi;
        for (int ky = 0; ky < 3; ++ky) {
            int yy = y + ky - 1; if (yy < 0 || yy >= Himg) continue;
            for (int kx = 0; kx < 3; ++kx) {
                int xx = x + kx - 1; if (xx < 0 || xx >= Wimg) continue;
                float v = ip[yy * Wimg + xx];
                const float* wp = wt + (size_t)ci * 9 + ky * 3 + kx;
#pragma unroll
                for (int co = 0; co < 16; ++co)
                    acc[co] += v * wp[(size_t)co * CIN * 9];
            }
        }
    }
#pragma unroll
    for (int co = 0; co < 16; ++co) {
        float v = acc[co]; v = v >= 0.f ? v : 0.1f * v;
        out[((size_t)b * 16 + co) * HWi + rp] = v;
    }
}

// ---------------------------------------------------------------------------
// Build the padded NHWC bf16 combined tensor [B][258][258][288]
// channels: feat1(32) feat2(32) corr(169) fr1(16) fr2(16) t(1) pad(22)
// ---------------------------------------------------------------------------
__global__ void pack_combined_k(const float* __restrict__ feat1, const float* __restrict__ feat2,
                                const float* __restrict__ corr, const float* __restrict__ fr1,
                                const float* __restrict__ fr2, __bf16* __restrict__ dst) {
    int idx = blockIdx.x * blockDim.x + threadIdx.x;
    if (idx >= BATCHN * HPc * WPc) return;
    int b = idx / (HPc * WPc);
    int rp = idx % (HPc * WPc);
    int py = rp / WPc, px = rp % WPc;
    __bf16* o = dst + (size_t)idx * CPc;
    if (py == 0 || py == HPc - 1 || px == 0 || px == WPc - 1) {
        for (int c = 0; c < CPc; ++c) o[c] = f2bf(0.f);
        return;
    }
    size_t r = (size_t)(py - 1) * Wimg + (px - 1);
    int c = 0;
    for (int i = 0; i < 32;  ++i) o[c++] = f2bf(feat1[((size_t)b * 32  + i) * HWi + r]);
    for (int i = 0; i < 32;  ++i) o[c++] = f2bf(feat2[((size_t)b * 32  + i) * HWi + r]);
    for (int i = 0; i < 169; ++i) o[c++] = f2bf(corr [((size_t)b * 169 + i) * HWi + r]);
    for (int i = 0; i < 16;  ++i) o[c++] = f2bf(fr1  [((size_t)b * 16  + i) * HWi + r]);
    for (int i = 0; i < 16;  ++i) o[c++] = f2bf(fr2  [((size_t)b * 16  + i) * HWi + r]);
    o[c++] = f2bf(0.5f);                           // TIMESTEP map
    while (c < CPc) o[c++] = f2bf(0.f);
}

// ---------------------------------------------------------------------------
// Implicit-GEMM conv3x3 with WMMA bf16.
//   src : NHWC padded bf16, CP channels (multiple of 32)
//   wpack: per-lane B fragments as produced by pack_weights_k
//   Block = 256 threads = 8 waves; block tile M=128 pixels x N=NTG*16 couts.
//   B tiles staged into double-buffered LDS with GLOBAL_LOAD_ASYNC_TO_LDS_B128
//   (ASYNCcnt), one workgroup barrier per K-chunk.
//   K loop is (unrolled 9 taps) x (channel chunks) -> no divisions in the
//   hot loop; A pointer and staging pointer advance linearly.
//   DSTM=0: write NHWC padded bf16 (with optional lrelu)
//   DSTM=1: write NCHW f32 (co < Nout only)
// ---------------------------------------------------------------------------
template<int ACT, int DSTM, int NTG>
__global__ __launch_bounds__(256)
void conv_gemm_k(const __bf16* __restrict__ src, int CP, int cpc, int nChunks,
                 const __bf16* __restrict__ wpack, const float* __restrict__ bias,
                 int Nout, int Ngroups, __bf16* __restrict__ dstA, int DCP,
                 float* __restrict__ dstB) {
    __shared__ uint4 ldsB[2 * NTG * 64];
    const int tid  = threadIdx.x;
    const int lane = tid & 31;
    const int wv   = tid >> 5;
    const int b      = blockIdx.z / Ngroups;
    const int ngBase = (blockIdx.z % Ngroups) * NTG;
    const int y      = blockIdx.y;
    const int xbase  = blockIdx.x * 128 + wv * 16;
    const int m    = lane & 15;
    const int hsel = lane >> 4;
    const int x    = xbase + m;
    const uint4* wp4 = reinterpret_cast<const uint4*>(wpack);

    // Async-stage setup: thread t owns one 16B fragment slot per chunk.
    const bool stager = (tid < NTG * 64);
    const int  snt  = tid >> 6;
    const int  srem = tid & 63;
    // global fragment pointer for chunk 0; advances by 64 uint4 (1KB) per chunk
    const uint4* sgp0 = wp4 + ((((size_t)(ngBase + snt) * nChunks) * 32 + (srem >> 1)) * 2
                               + (srem & 1));
    // LDS byte address: flat shared address truncated to 32 bits == LDS offset
    const unsigned ldsDst0 = (unsigned)(uintptr_t)(&ldsB[0]) + (unsigned)tid * 16u;
    const unsigned bufStride = (unsigned)(NTG * 64 * 16);

    v8f acc[NTG] = {};

    // prologue: stage chunk 0 into buffer 0
    if (stager) {
        asm volatile("global_load_async_to_lds_b128 %0, %1, off"
                     :: "v"(ldsDst0), "v"(sgp0) : "memory");
    }

    int c = 0;
    const uint4* sgpNext = sgp0 + 64;     // chunk c+1 staging source
#pragma unroll
    for (int pos = 0; pos < 9; ++pos) {
        const int ky = pos / 3, kx = pos % 3;   // compile-time after unroll
        const __bf16* ap = src + (((size_t)(b * HPc + (y + ky)) * WPc) + (x + kx)) * CP
                               + hsel * 8;
        for (int ci = 0; ci < cpc; ++ci) {
            Frag32 A;
            A.u[0] = *reinterpret_cast<const uint4*>(ap);
            A.u[1] = *reinterpret_cast<const uint4*>(ap + 16);
            ap += 32;

            // my async LDS writes complete -> barrier makes everyone's visible
            asm volatile("s_wait_asynccnt 0x0" ::: "memory");
            __syncthreads();

            // overlap: kick off next chunk's staging into the other buffer
            if (c + 1 < nChunks && stager) {
                unsigned dst = ldsDst0 + (unsigned)((c + 1) & 1) * bufStride;
                asm volatile("global_load_async_to_lds_b128 %0, %1, off"
                             :: "v"(dst), "v"(sgpNext) : "memory");
            }
            sgpNext += 64;

            const uint4* lp = &ldsB[(c & 1) * (NTG * 64)];
#pragma unroll
            for (int nt = 0; nt < NTG; ++nt) {
                Frag32 Bf;
                Bf.u[0] = lp[nt * 64 + lane * 2];
                Bf.u[1] = lp[nt * 64 + lane * 2 + 1];
                acc[nt] = __builtin_amdgcn_wmma_f32_16x16x32_bf16(
                    false, A.v, false, Bf.v, (short)0, acc[nt], false, false);
            }
            ++c;
        }
    }

    // epilogue: C/D layout lane l, reg r -> M = r + 8*(l>>4), N = l&15
    const int n = lane & 15;
#pragma unroll
    for (int nt = 0; nt < NTG; ++nt) {
        const int co = (ngBase + nt) * 16 + n;
        const float bv = (co < Nout) ? bias[co] : 0.f;
        v8f a = acc[nt];
#pragma unroll
        for (int r = 0; r < 8; ++r) {
            const int px = xbase + r + 8 * hsel;
            float v = a[r] + bv;
            if (ACT) v = v >= 0.f ? v : 0.1f * v;
            if (DSTM == 0) {
                dstA[((size_t)(b * HPc + (y + 1)) * WPc + (px + 1)) * DCP + co] = f2bf(v);
            } else {
                if (co < Nout)
                    dstB[(((size_t)b * Nout + co) * Himg + y) * Wimg + px] = v;
            }
        }
    }
}

// ---------------------------------------------------------------------------
// Softmax over the 25 kernel-weight channels (NCHW f32 -> NCHW f32)
// ---------------------------------------------------------------------------
__global__ void softmax25_k(const float* __restrict__ pre, float* __restrict__ dst) {
    int idx = blockIdx.x * blockDim.x + threadIdx.x;
    if (idx >= BATCHN * HWi) return;
    int b = idx / HWi;
    size_t r = idx % HWi;
    float vals[25];
    float mx = -3.4e38f;
#pragma unroll
    for (int k = 0; k < 25; ++k) {
        vals[k] = pre[((size_t)b * 25 + k) * HWi + r];
        mx = fmaxf(mx, vals[k]);
    }
    float s = 0.f;
#pragma unroll
    for (int k = 0; k < 25; ++k) { vals[k] = expf(vals[k] - mx); s += vals[k]; }
    float inv = 1.f / s;
#pragma unroll
    for (int k = 0; k < 25; ++k)
        dst[((size_t)b * 25 + k) * HWi + r] = vals[k] * inv;
}

// ---------------------------------------------------------------------------
// AdaCoF bicubic sampling: per pixel, 25 offset taps x 4x4 cubic x 3 channels
// ---------------------------------------------------------------------------
__device__ __forceinline__ void cubw(float t, float w[4]) {
    const float A = -0.75f;
    w[1] = ((A + 2.f) * t - (A + 3.f)) * t * t + 1.f;
    float s = 1.f - t;
    w[2] = ((A + 2.f) * s - (A + 3.f)) * s * s + 1.f;
    float u = t + 1.f;
    w[0] = ((A * u - 5.f * A) * u + 8.f * A) * u - 4.f * A;
    float v = 2.f - t;
    w[3] = ((A * v - 5.f * A) * v + 8.f * A) * v - 4.f * A;
}

__global__ void adacof_k(const float* __restrict__ img, const float* __restrict__ off,
                         const float* __restrict__ wtn, float* __restrict__ samp) {
    int idx = blockIdx.x * blockDim.x + threadIdx.x;
    if (idx >= BATCHN * HWi) return;
    int b = idx / HWi;
    int rp = idx % HWi;
    int y = rp / Wimg, x = rp % Wimg;
    float a0 = 0.f, a1 = 0.f, a2 = 0.f;
    const float* i0 = img + (size_t)b * 3 * HWi;
    for (int k = 0; k < 25; ++k) {
        float dx = off[((size_t)b * 50 + 2 * k    ) * HWi + rp];
        float dy = off[((size_t)b * 50 + 2 * k + 1) * HWi + rp];
        float wk = wtn[((size_t)b * 25 + k) * HWi + rp];
        float sx = (float)x + dx, sy = (float)y + dy;
        float fx = floorf(sx), fy = floorf(sy);
        float wx[4], wy[4];
        cubw(sx - fx, wx);
        cubw(sy - fy, wy);
        int xi[4], yi[4];
#pragma unroll
        for (int j = 0; j < 4; ++j) {
            xi[j] = (int)fminf(fmaxf(fx + (float)(j - 1), 0.f), (float)(Wimg - 1));
            yi[j] = (int)fminf(fmaxf(fy + (float)(j - 1), 0.f), (float)(Himg - 1));
        }
#pragma unroll
        for (int i = 0; i < 4; ++i) {
#pragma unroll
            for (int j = 0; j < 4; ++j) {
                float ww = wk * wy[i] * wx[j];
                size_t p = (size_t)yi[i] * Wimg + xi[j];
                a0 += ww * i0[p];
                a1 += ww * i0[HWi + p];
                a2 += ww * i0[2 * (size_t)HWi + p];
            }
        }
    }
    samp[((size_t)b * 3 + 0) * HWi + rp] = a0;
    samp[((size_t)b * 3 + 1) * HWi + rp] = a1;
    samp[((size_t)b * 3 + 2) * HWi + rp] = a2;
}

// ---------------------------------------------------------------------------
// Final blend: sigmoid(blend_pre) mix of the two samples
// ---------------------------------------------------------------------------
__global__ void final_k(const float* __restrict__ blpre, const float* __restrict__ s1,
                        const float* __restrict__ s2, float* __restrict__ outimg,
                        float* __restrict__ outblend) {
    int idx = blockIdx.x * blockDim.x + threadIdx.x;
    if (idx >= BATCHN * HWi) return;
    int b = idx / HWi;
    size_t rp = idx % HWi;
    float bl = 1.f / (1.f + expf(-blpre[idx]));
    outblend[idx] = bl;
#pragma unroll
    for (int c = 0; c < 3; ++c) {
        size_t o = ((size_t)b * 3 + c) * HWi + rp;
        outimg[o] = bl * s1[o] + (1.f - bl) * s2[o];
    }
}

// ---------------------------------------------------------------------------
// Host-side orchestration
// ---------------------------------------------------------------------------
extern "C" void kernel_launch(void* const* d_in, const int* in_sizes, int n_in,
                              void* d_out, int out_size, void* d_ws, size_t ws_size,
                              hipStream_t stream) {
    (void)in_sizes; (void)n_in; (void)out_size; (void)ws_size;
    const float* frame1 = (const float*)d_in[0];
    const float* frame2 = (const float*)d_in[1];
    const float* feat1  = (const float*)d_in[2];
    const float* feat2  = (const float*)d_in[3];
    const float* corr   = (const float*)d_in[4];
    const float* fr_w1  = (const float*)d_in[5];
    const float* fr_b1  = (const float*)d_in[6];
    const float* fr_w2  = (const float*)d_in[7];
    const float* fr_b2  = (const float*)d_in[8];
    const float* kp_w1[2] = {(const float*)d_in[9],  (const float*)d_in[19]};
    const float* kp_b1[2] = {(const float*)d_in[10], (const float*)d_in[20]};
    const float* kp_w2[2] = {(const float*)d_in[11], (const float*)d_in[21]};
    const float* kp_b2[2] = {(const float*)d_in[12], (const float*)d_in[22]};
    const float* kp_w3[2] = {(const float*)d_in[13], (const float*)d_in[23]};
    const float* kp_b3[2] = {(const float*)d_in[14], (const float*)d_in[24]};
    const float* kp_ow[2] = {(const float*)d_in[15], (const float*)d_in[25]};
    const float* kp_ob[2] = {(const float*)d_in[16], (const float*)d_in[26]};
    const float* kp_ww[2] = {(const float*)d_in[17], (const float*)d_in[27]};
    const float* kp_wb[2] = {(const float*)d_in[18], (const float*)d_in[28]};
    const float* bl_w1 = (const float*)d_in[29];
    const float* bl_b1 = (const float*)d_in[30];
    const float* bl_w2 = (const float*)d_in[31];
    const float* bl_b2 = (const float*)d_in[32];

    float* outp = (float*)d_out;
    float* out_img   = outp;                         // [2,3,256,256]
    float* out_blend = outp + 393216;                // [2,1,256,256]
    float* out_off[2] = { outp + 524288,  outp + 7077888  };   // [2,50,256,256] x2
    float* out_wt[2]  = { outp + 13631488, outp + 16908288 };  // [2,25,256,256] x2

    char* ws = (char*)d_ws;
    size_t woff = 0;
    auto alloc = [&](size_t bytes) -> void* {
        void* p = ws + woff;
        woff = (woff + bytes + 255) & ~((size_t)255);
        return p;
    };
    const size_t planes = (size_t)BATCHN * HPc * WPc;
    __bf16* combined = (__bf16*)alloc(planes * CPc * 2);
    const size_t szActA = planes * 256 * 2;
    __bf16* actA   = (__bf16*)alloc(szActA);
    __bf16* actB   = (__bf16*)alloc(szActA);
    const size_t szAct128 = planes * 128 * 2;
    __bf16* act128 = (__bf16*)alloc(szAct128);
    const size_t szAct64 = planes * 64 * 2;
    __bf16* act64  = (__bf16*)alloc(szAct64);
    float* frTmp = (float*)alloc((size_t)BATCHN * 16 * HWi * 4);
    float* fr1   = (float*)alloc((size_t)BATCHN * 16 * HWi * 4);
    float* fr2   = (float*)alloc((size_t)BATCHN * 16 * HWi * 4);
    float* wsW   = (float*)alloc((size_t)BATCHN * 25 * HWi * 4);
    float* wsBl  = (float*)alloc((size_t)BATCHN * HWi * 4);
    float* s1    = (float*)alloc((size_t)BATCHN * 3 * HWi * 4);
    float* s2    = (float*)alloc((size_t)BATCHN * 3 * HWi * 4);
    auto wpAlloc = [&](int ntile, int nc) { return (__bf16*)alloc((size_t)ntile * nc * 1024); };
    __bf16* wpL1[2] = { wpAlloc(16, 81), wpAlloc(16, 81) };
    __bf16* wpL2[2] = { wpAlloc(16, 72), wpAlloc(16, 72) };
    __bf16* wpL3[2] = { wpAlloc(8, 72),  wpAlloc(8, 72)  };
    __bf16* wpOW[2] = { wpAlloc(4, 36),  wpAlloc(4, 36)  };
    __bf16* wpWW[2] = { wpAlloc(2, 36),  wpAlloc(2, 36)  };
    __bf16* wpB1 = wpAlloc(4, 81);
    __bf16* wpB2 = wpAlloc(1, 18);

    // zero activation buffers so padded borders are zeros
    hipMemsetAsync(actA,   0, szActA,   stream);
    hipMemsetAsync(actB,   0, szActA,   stream);
    hipMemsetAsync(act128, 0, szAct128, stream);
    hipMemsetAsync(act64,  0, szAct64,  stream);

    auto packW = [&](const float* Wsrc, __bf16* dst, int O, int I, int CP, int nc, int nt) {
        int total = nt * nc * 32;
        pack_weights_k<<<(total + 255) / 256, 256, 0, stream>>>(Wsrc, dst, O, I, CP, nc, total);
    };
    for (int p = 0; p < 2; ++p) {
        packW(kp_w1[p], wpL1[p], 256, 266, 288, 81, 16);
        packW(kp_w2[p], wpL2[p], 256, 256, 256, 72, 16);
        packW(kp_w3[p], wpL3[p], 128, 256, 256, 72, 8);
        packW(kp_ow[p], wpOW[p], 50, 128, 128, 36, 4);
        packW(kp_ww[p], wpWW[p], 25, 128, 128, 36, 2);
    }
    packW(bl_w1, wpB1, 64, 266, 288, 81, 4);
    packW(bl_w2, wpB2, 1, 64, 64, 18, 1);

    const int pixBlocks = (BATCHN * HWi + 255) / 256;
    frconv_k<3><<<pixBlocks, 256, 0, stream>>>(frame1, fr_w1, fr_b1, frTmp);
    frconv_k<16><<<pixBlocks, 256, 0, stream>>>(frTmp, fr_w2, fr_b2, fr1);
    frconv_k<3><<<pixBlocks, 256, 0, stream>>>(frame2, fr_w1, fr_b1, frTmp);
    frconv_k<16><<<pixBlocks, 256, 0, stream>>>(frTmp, fr_w2, fr_b2, fr2);

    const int padBlocks = (BATCHN * HPc * WPc + 255) / 256;
    pack_combined_k<<<padBlocks, 256, 0, stream>>>(feat1, feat2, corr, fr1, fr2, combined);

    dim3 blk(256);
    for (int p = 0; p < 2; ++p) {
        conv_gemm_k<1, 0, 4><<<dim3(2, 256, BATCHN * 4), blk, 0, stream>>>(
            combined, 288, 9, 81, wpL1[p], kp_b1[p], 256, 4, actA, 256, nullptr);
        conv_gemm_k<1, 0, 4><<<dim3(2, 256, BATCHN * 4), blk, 0, stream>>>(
            actA, 256, 8, 72, wpL2[p], kp_b2[p], 256, 4, actB, 256, nullptr);
        conv_gemm_k<1, 0, 4><<<dim3(2, 256, BATCHN * 2), blk, 0, stream>>>(
            actB, 256, 8, 72, wpL3[p], kp_b3[p], 128, 2, act128, 128, nullptr);
        conv_gemm_k<0, 1, 4><<<dim3(2, 256, BATCHN * 1), blk, 0, stream>>>(
            act128, 128, 4, 36, wpOW[p], kp_ob[p], 50, 1, nullptr, 0, out_off[p]);
        conv_gemm_k<0, 1, 2><<<dim3(2, 256, BATCHN * 1), blk, 0, stream>>>(
            act128, 128, 4, 36, wpWW[p], kp_wb[p], 25, 1, nullptr, 0, wsW);
        softmax25_k<<<pixBlocks, 256, 0, stream>>>(wsW, out_wt[p]);
    }

    conv_gemm_k<1, 0, 4><<<dim3(2, 256, BATCHN * 1), blk, 0, stream>>>(
        combined, 288, 9, 81, wpB1, bl_b1, 64, 1, act64, 64, nullptr);
    conv_gemm_k<0, 1, 1><<<dim3(2, 256, BATCHN * 1), blk, 0, stream>>>(
        act64, 64, 2, 18, wpB2, bl_b2, 1, 1, nullptr, 0, wsBl);

    adacof_k<<<pixBlocks, 256, 0, stream>>>(frame1, out_off[0], out_wt[0], s1);
    adacof_k<<<pixBlocks, 256, 0, stream>>>(frame2, out_off[1], out_wt[1], s2);
    final_k<<<pixBlocks, 256, 0, stream>>>(wsBl, s1, s2, out_img, out_blend);
}
